// LGEB_17257178595649
// MI455X (gfx1250) — matrix operand order repaired and checked by reading, not compiled, not run on hardware
//
#include <hip/hip_runtime.h>
#include <hip/hip_bf16.h>
#include <math.h>

// ---- problem constants (match reference) ----
#define B_   32
#define N_   128
#define E_   16384
#define NIc  72
#define NHc  72
#define NOc  72
#define NAc  8
#define K1   160   // K padded for 146 (edge feat) and 152 (node cat)
#define K2   96    // K padded for 72-wide intermediates (3 x 32)
#define CP   80    // padded output columns for 72 (5 x 16)
#define TME  128   // rows per edge block tile
#define NTE  40    // (TME/16) * (CP/16)
#define TMN  64    // rows per node block tile
#define NTN  20    // (TMN/16) * (CP/16)
#define EPSc 1e-5f

typedef __attribute__((ext_vector_type(16))) __bf16       v16bf;
typedef __attribute__((ext_vector_type(8)))  float        v8f;
typedef __attribute__((ext_vector_type(4)))  unsigned int u32x4;

union BfPack { v16bf v; u32x4 q[2]; };

__device__ __forceinline__ unsigned short f2bf(float f) {
    unsigned u = __builtin_bit_cast(unsigned, f);
    unsigned r = 0x7FFFu + ((u >> 16) & 1u);       // round-to-nearest-even
    return (unsigned short)((u + r) >> 16);
}
__device__ __forceinline__ float bf2f(unsigned short s) {
    return __builtin_bit_cast(float, ((unsigned)s) << 16);
}
__device__ __forceinline__ float psi_f(float v) {
    return copysignf(log1pf(fabsf(v)), v);
}

// 16-byte vectorized global->LDS staging copy (n_ush multiple of 8, both aligned)
__device__ __forceinline__ void stage16(unsigned short* dst, const unsigned short* src,
                                        int n_ush, int tid) {
    const u32x4* s = (const u32x4*)src;
    u32x4* d = (u32x4*)dst;
    for (int i = tid; i < (n_ush >> 3); i += 256) d[i] = s[i];
}
// global_prefetch_b8 hint over a byte range
__device__ __forceinline__ void prefetch_bytes(const void* p, int bytes, int tid) {
    const char* c = (const char*)p;
    for (int o = tid * 128; o < bytes; o += 256 * 128) __builtin_prefetch(c + o, 0, 1);
}

// One 16x16 f32 output tile of A[rows x K] * B (B stored transposed: Bt[col][K]).
// A,Bt are bf16-in-ushort LDS arrays. nk = K/32 tiles.
__device__ __forceinline__ v8f wmma_tile(const unsigned short* A, int lda,
                                         const unsigned short* Bt, int ldb,
                                         int m0, int n0, int nk, int lane) {
    const int l15  = lane & 15;
    const int half = (lane >> 4) & 1;
    const unsigned short* ar = A + (m0 + l15) * lda;
    const unsigned short* bc = Bt + (n0 + l15) * ldb;
    v8f acc = {};
    for (int kt = 0; kt < nk; ++kt) {
        BfPack a, b;
        const int k = kt * 32;
        a.q[0] = *(const u32x4*)(ar + k + 8 * half);        // A: K = 8h..8h+7, +16
        a.q[1] = *(const u32x4*)(ar + k + 16 + 8 * half);
        b.q[0] = *(const u32x4*)(bc + k + 8 * half);        // B: K = 8h..8h+7, +16
        b.q[1] = *(const u32x4*)(bc + k + 16 + 8 * half);
        acc = __builtin_amdgcn_wmma_f32_16x16x32_bf16(false, a.v, false, b.v,
                                                      (short)0, acc, false, false);
    }
    return acc;
}

// ---------------- stage 0a: zero workspace accumulators ----------------
__global__ void k_zero(float* __restrict__ p, int n) {
    int i = blockIdx.x * 256 + threadIdx.x;
    if (i < n) p[i] = 0.f;
}

// ---------------- stage 0b: pre-convert a weight matrix to bf16, transposed+padded ----
// W is [kdim x ndim] row-major f32; dst is [CP x ldk] bf16 with dst[n*ldk+k] = W[k][n]
__global__ void k_conv_wt(const float* __restrict__ W, unsigned short* __restrict__ dst,
                          int kdim, int ndim, int ldk) {
    int idx = blockIdx.x * 256 + threadIdx.x;
    if (idx < CP * ldk) {
        int n = idx / ldk, k = idx % ldk;
        dst[idx] = (n < ndim && k < kdim) ? f2bf(W[k * ndim + n]) : (unsigned short)0;
    }
}
// pre-convert h to bf16 (flat copy)
__global__ void k_conv_h(const float* __restrict__ src, unsigned short* __restrict__ dst, int n) {
    int i = blockIdx.x * 256 + threadIdx.x;
    if (i < n) dst[i] = f2bf(src[i]);
}

// ---------------- stage 1: edge GEMM1 + BN stats ----------------
__global__ void k_edge_stats(const unsigned short* __restrict__ hbf, const float* __restrict__ x,
                             const int* __restrict__ ei, const int* __restrict__ ej,
                             const unsigned short* __restrict__ wtE1g,
                             float* __restrict__ sumE, float* __restrict__ sumsqE) {
    extern __shared__ char smem[];
    unsigned short* feat  = (unsigned short*)smem;   // TME*K1
    unsigned short* wt    = feat + TME * K1;         // CP*K1
    float* colsum = (float*)(wt + CP * K1);          // CP
    float* colsq  = colsum + CP;                     // CP
    int*   ii     = (int*)(colsq + CP);
    int*   jj     = ii + TME;

    const int tid = threadIdx.x;
    const int g0  = blockIdx.x * TME;
    const int bb  = g0 / E_;                         // TME divides E_, block-uniform
    const unsigned short* hb = hbf + (size_t)bb * N_ * NIc;

    prefetch_bytes(hb, N_ * NIc * 2, tid);
    stage16(wt, wtE1g, CP * K1, tid);
    if (tid < 2 * CP) colsum[tid] = 0.f;             // colsum+colsq contiguous
    if (tid < TME) {
        int g = g0 + tid;
        int i = ei[g], j = ej[g];
        ii[tid] = i; jj[tid] = j;
        const float* xi = x + (bb * N_ + i) * 4;
        const float* xj = x + (bb * N_ + j) * 4;
        float d0 = xi[0]-xj[0], d1 = xi[1]-xj[1], d2 = xi[2]-xj[2], d3 = xi[3]-xj[3];
        float nrm = psi_f(d0*d0 - d1*d1 - d2*d2 - d3*d3);
        float dot = psi_f(xi[0]*xj[0] - xi[1]*xj[1] - xi[2]*xj[2] - xi[3]*xj[3]);
        unsigned short* fr = feat + tid * K1;
        fr[144] = f2bf(nrm); fr[145] = f2bf(dot);
        for (int c = 146; c < K1; ++c) fr[c] = 0;
    }
    __syncthreads();
    for (int idx = tid; idx < TME * 2 * NIc; idx += 256) {
        int r = idx / (2 * NIc), c = idx % (2 * NIc);
        int node = (c < NIc) ? ii[r] : jj[r];
        int cc   = (c < NIc) ? c : c - NIc;
        feat[r * K1 + c] = hb[node * NIc + cc];
    }
    __syncthreads();

    const int wid = tid >> 5, lane = tid & 31;
    for (int t = wid; t < NTE; t += 8) {
        int mt = t / 5, nt = t % 5;
        v8f acc = wmma_tile(feat, K1, wt, K1, mt * 16, nt * 16, K1 / 32, lane);
        float s = 0.f, ss = 0.f;
        for (int r = 0; r < 8; ++r) { float v = acc[r]; s += v; ss += v * v; }
        s  += __shfl_xor(s, 16, 32);
        ss += __shfl_xor(ss, 16, 32);
        if (lane < 16) {
            atomicAdd(&colsum[nt * 16 + lane], s);
            atomicAdd(&colsq [nt * 16 + lane], ss);
        }
    }
    __syncthreads();
    if (tid < NHc) {
        atomicAdd(&sumE[tid],   colsum[tid]);
        atomicAdd(&sumsqE[tid], colsq[tid]);
    }
}

// ---------------- stage 2/5: finalize BN scale/shift ----------------
__global__ void k_bn_finalize(const float* __restrict__ sum, const float* __restrict__ sumsq,
                              const float* __restrict__ g, const float* __restrict__ be,
                              float* __restrict__ scale, float* __restrict__ shift,
                              float count) {
    int t = threadIdx.x;
    if (t < CP) {
        if (t < NHc) {
            float mu  = sum[t] / count;
            float var = sumsq[t] / count - mu * mu;
            float sc  = g[t] * rsqrtf(var + EPSc);
            scale[t]  = sc;
            shift[t]  = be[t] - mu * sc;
        } else { scale[t] = 0.f; shift[t] = 0.f; }
    }
}

// ---------------- stage 3: fused edge pipeline ----------------
__global__ void k_edge_main(const unsigned short* __restrict__ hbf, const float* __restrict__ x,
                            const int* __restrict__ ei, const int* __restrict__ ej,
                            const unsigned short* __restrict__ wtE1g,
                            const unsigned short* __restrict__ wtE2g,
                            const unsigned short* __restrict__ wtX1g,
                            const float* __restrict__ be2g,
                            const float* __restrict__ Wm_g, const float* __restrict__ bm_g,
                            const float* __restrict__ bx1g, const float* __restrict__ Wx2g,
                            const float* __restrict__ scEg, const float* __restrict__ shEg,
                            float* __restrict__ out_m, float* __restrict__ magg,
                            float* __restrict__ agg, float* __restrict__ cnt) {
    extern __shared__ char smem[];
    unsigned short* bufA = (unsigned short*)smem;    // feat (K1) -> zA/mB (K2)
    unsigned short* wt   = bufA + TME * K1;          // CP*K1, reused with stride K2
    unsigned short* bufB = wt + CP * K1;             // TME*K2 : m_e -> phi_x hidden
    float* scE  = (float*)(bufB + TME * K2);         // CP
    float* shE  = scE + CP;
    float* bE2  = shE + CP;
    float* bX1  = bE2 + CP;
    float* Wm   = bX1 + CP;                          // NHc
    float* Wx2  = Wm + NHc;                          // NHc
    float* wrow = Wx2 + NHc;                         // TME
    float* xdf  = wrow + TME;                        // TME*4
    int*   ii   = (int*)(xdf + TME * 4);
    int*   jj   = ii + TME;

    const int tid  = threadIdx.x;
    const int wid  = tid >> 5, lane = tid & 31;
    const int l15  = lane & 15, half = (lane >> 4) & 1;
    const int g0   = blockIdx.x * TME;
    const int bb   = g0 / E_;
    const unsigned short* hb = hbf + (size_t)bb * N_ * NIc;

    prefetch_bytes(hb, N_ * NIc * 2, tid);
    stage16(wt, wtE1g, CP * K1, tid);
    if (tid < CP) {
        scE[tid] = scEg[tid];
        shE[tid] = shEg[tid];
        bE2[tid] = (tid < NHc) ? be2g[tid] : 0.f;
        bX1[tid] = (tid < NHc) ? bx1g[tid] : 0.f;
    }
    if (tid < NHc) { Wm[tid] = Wm_g[tid]; Wx2[tid] = Wx2g[tid]; }
    if (tid < TME) {
        int g = g0 + tid;
        int i = ei[g], j = ej[g];
        ii[tid] = i; jj[tid] = j;
        const float* xi = x + (bb * N_ + i) * 4;
        const float* xj = x + (bb * N_ + j) * 4;
        float d0 = xi[0]-xj[0], d1 = xi[1]-xj[1], d2 = xi[2]-xj[2], d3 = xi[3]-xj[3];
        xdf[tid*4+0] = d0; xdf[tid*4+1] = d1; xdf[tid*4+2] = d2; xdf[tid*4+3] = d3;
        float nrm = psi_f(d0*d0 - d1*d1 - d2*d2 - d3*d3);
        float dot = psi_f(xi[0]*xj[0] - xi[1]*xj[1] - xi[2]*xj[2] - xi[3]*xj[3]);
        unsigned short* fr = bufA + tid * K1;
        fr[144] = f2bf(nrm); fr[145] = f2bf(dot);
        for (int c = 146; c < K1; ++c) fr[c] = 0;
    }
    __syncthreads();
    for (int idx = tid; idx < TME * 2 * NIc; idx += 256) {
        int r = idx / (2 * NIc), c = idx % (2 * NIc);
        int node = (c < NIc) ? ii[r] : jj[r];
        int cc   = (c < NIc) ? c : c - NIc;
        bufA[r * K1 + c] = hb[node * NIc + cc];
    }
    __syncthreads();

    // GEMM1: hold results in registers across the barrier, then BN+ReLU -> zA
    float accs[NTE / 8][8];
    {
        int q = 0;
        for (int t = wid; t < NTE; t += 8, ++q) {
            v8f a = wmma_tile(bufA, K1, wt, K1, (t / 5) * 16, (t % 5) * 16, K1 / 32, lane);
            for (int r = 0; r < 8; ++r) accs[q][r] = a[r];
        }
    }
    __syncthreads();
    {
        int q = 0;
        for (int t = wid; t < NTE; t += 8, ++q) {
            int mt = t / 5, nt = t % 5, col = nt * 16 + l15;
            int row0 = mt * 16 + half * 8;
            for (int r = 0; r < 8; ++r) {
                float v = fmaxf(accs[q][r] * scE[col] + shE[col], 0.f);
                bufA[(row0 + r) * K2 + col] = f2bf(v);
            }
        }
    }
    for (int idx = tid; idx < TME * 16; idx += 256) {         // zero K2 pad cols 80..95
        int r = idx >> 4, c = CP + (idx & 15);
        bufA[r * K2 + c] = 0; bufB[r * K2 + c] = 0;
    }
    stage16(wt, wtE2g, CP * K2, tid);                         // wt <- We2^T (bf16 image)
    __syncthreads();

    // GEMM2: m_e = relu(zA @ We2 + b)
    for (int t = wid; t < NTE; t += 8) {
        int mt = t / 5, nt = t % 5, col = nt * 16 + l15;
        v8f a = wmma_tile(bufA, K2, wt, K2, mt * 16, nt * 16, K2 / 32, lane);
        int row0 = mt * 16 + half * 8;
        for (int r = 0; r < 8; ++r)
            bufB[(row0 + r) * K2 + col] = f2bf(fmaxf(a[r] + bE2[col], 0.f));
    }
    __syncthreads();

    {   // sigmoid gate: 2 lanes per row, shuffle-combine
        int r = tid >> 1, part = tid & 1;
        float s = 0.f;
        for (int c = part * 36; c < part * 36 + 36; ++c)
            s += bf2f(bufB[r * K2 + c]) * Wm[c];
        s += __shfl_xor(s, 1, 32);
        if (part == 0) wrow[r] = 1.f / (1.f + __expf(-(s + bm_g[0])));
    }
    __syncthreads();

    // m = m_e * w : emit output, scatter magg, stage mB for GEMM3
    for (int idx = tid; idx < TME * NHc; idx += 256) {
        int r = idx / NHc, c = idx % NHc;
        float mv = bf2f(bufB[r * K2 + c]) * wrow[r];
        out_m[(size_t)(g0 + r) * NHc + c] = mv;
        atomicAdd(&magg[(bb * N_ + ii[r]) * NHc + c], mv);
        bufA[r * K2 + c] = f2bf(mv);
    }
    stage16(wt, wtX1g, CP * K2, tid);                         // wt <- Wx1^T
    __syncthreads();

    // GEMM3: t = relu(mB @ Wx1 + b)
    for (int t = wid; t < NTE; t += 8) {
        int mt = t / 5, nt = t % 5, col = nt * 16 + l15;
        v8f a = wmma_tile(bufA, K2, wt, K2, mt * 16, nt * 16, K2 / 32, lane);
        int row0 = mt * 16 + half * 8;
        for (int r = 0; r < 8; ++r)
            bufB[(row0 + r) * K2 + col] = f2bf(fmaxf(a[r] + bX1[col], 0.f));
    }
    __syncthreads();

    {   // phi_x scalar + scatters: 2 lanes per row
        int r = tid >> 1, part = tid & 1;
        float p = 0.f;
        for (int c = part * 36; c < part * 36 + 36; ++c)
            p += bf2f(bufB[r * K2 + c]) * Wx2[c];
        p += __shfl_xor(p, 1, 32);
        if (part == 0) {
            int node = bb * N_ + ii[r];
            for (int d = 0; d < 4; ++d) {
                float tr = fminf(fmaxf(xdf[r * 4 + d] * p, -100.f), 100.f);
                atomicAdd(&agg[node * 4 + d], tr);
            }
            atomicAdd(&cnt[node], 1.f);
        }
    }
}

// ---------------- stage 4: node GEMM1 + BN stats ----------------
__global__ void k_node_gemm1(const unsigned short* __restrict__ hbf,
                             const float* __restrict__ nattr,
                             const float* __restrict__ magg,
                             const unsigned short* __restrict__ wtH1g,
                             const float* __restrict__ bh1g,
                             float* __restrict__ zh1,
                             float* __restrict__ sumH, float* __restrict__ sumsqH) {
    extern __shared__ char smem[];
    unsigned short* cat = (unsigned short*)smem;     // TMN*K1
    unsigned short* wt  = cat + TMN * K1;            // CP*K1
    float* colsum = (float*)(wt + CP * K1);
    float* colsq  = colsum + CP;
    float* bh1    = colsq + CP;

    const int tid = threadIdx.x;
    const int r0  = blockIdx.x * TMN;                // over B*N rows

    stage16(wt, wtH1g, CP * K1, tid);
    if (tid < CP) { bh1[tid] = (tid < NHc) ? bh1g[tid] : 0.f; colsum[tid] = 0.f; colsq[tid] = 0.f; }
    for (int idx = tid; idx < TMN * K1; idx += 256) {
        int r = idx / K1, c = idx % K1;
        int rg = r0 + r;
        unsigned short v;
        if      (c < NIc)             v = hbf[(size_t)rg * NIc + c];
        else if (c < NIc + NHc)       v = f2bf(magg[rg * NHc + (c - NIc)]);
        else if (c < NIc + NHc + NAc) v = f2bf(nattr[rg * NAc + (c - NIc - NHc)]);
        else                          v = 0;
        cat[idx] = v;
    }
    __syncthreads();

    const int wid = tid >> 5, lane = tid & 31, l15 = lane & 15, half = (lane >> 4) & 1;
    for (int t = wid; t < NTN; t += 8) {
        int mt = t / 5, nt = t % 5, col = nt * 16 + l15;
        v8f a = wmma_tile(cat, K1, wt, K1, mt * 16, nt * 16, K1 / 32, lane);
        int row0 = mt * 16 + half * 8;
        float s = 0.f, ss = 0.f;
        for (int r = 0; r < 8; ++r) {
            float v = a[r] + bh1[col];
            if (col < NHc) zh1[(size_t)(r0 + row0 + r) * NHc + col] = v;
            s += v; ss += v * v;
        }
        s  += __shfl_xor(s, 16, 32);
        ss += __shfl_xor(ss, 16, 32);
        if (lane < 16) { atomicAdd(&colsum[nt * 16 + lane], s); atomicAdd(&colsq[nt * 16 + lane], ss); }
    }
    __syncthreads();
    if (tid < NHc) { atomicAdd(&sumH[tid], colsum[tid]); atomicAdd(&sumsqH[tid], colsq[tid]); }
}

// ---------------- stage 6: node GEMM2 + residual + x_out ----------------
__global__ void k_node_final(const float* __restrict__ zh1,
                             const float* __restrict__ scHg, const float* __restrict__ shHg,
                             const unsigned short* __restrict__ wtH2g,
                             const float* __restrict__ bh2g,
                             const float* __restrict__ h, const float* __restrict__ x,
                             const float* __restrict__ agg, const float* __restrict__ cnt,
                             float* __restrict__ out_h, float* __restrict__ out_x) {
    extern __shared__ char smem[];
    unsigned short* zn = (unsigned short*)smem;      // TMN*K2
    unsigned short* wt = zn + TMN * K2;              // CP*K2
    float* scH = (float*)(wt + CP * K2);
    float* shH = scH + CP;
    float* bh2 = shH + CP;

    const int tid = threadIdx.x;
    const int r0  = blockIdx.x * TMN;
    stage16(wt, wtH2g, CP * K2, tid);
    if (tid < CP) { scH[tid] = scHg[tid]; shH[tid] = shHg[tid]; bh2[tid] = (tid < NOc) ? bh2g[tid] : 0.f; }
    __syncthreads();                                 // scH needed for zn build
    for (int idx = tid; idx < TMN * K2; idx += 256) {
        int r = idx / K2, c = idx % K2;
        float v = 0.f;
        if (c < NHc) v = fmaxf(zh1[(size_t)(r0 + r) * NHc + c] * scH[c] + shH[c], 0.f);
        zn[idx] = f2bf(v);
    }
    __syncthreads();

    const int wid = tid >> 5, lane = tid & 31, l15 = lane & 15, half = (lane >> 4) & 1;
    for (int t = wid; t < NTN; t += 8) {
        int mt = t / 5, nt = t % 5, col = nt * 16 + l15;
        v8f a = wmma_tile(zn, K2, wt, K2, mt * 16, nt * 16, K2 / 32, lane);
        int row0 = mt * 16 + half * 8;
        if (col < NOc)
            for (int r = 0; r < 8; ++r) {
                size_t o = (size_t)(r0 + row0 + r) * NOc + col;
                out_h[o] = a[r] + bh2[col] + h[o];
            }
    }
    if (tid < TMN) {
        int rg = r0 + tid;
        float cn = fmaxf(cnt[rg], 1.f);
        for (int d = 0; d < 4; ++d)
            out_x[rg * 4 + d] = x[rg * 4 + d] + (agg[rg * 4 + d] / cn);   // C_WEIGHT = 1
    }
}

// ---------------- host launch ----------------
extern "C" void kernel_launch(void* const* d_in, const int* in_sizes, int n_in,
                              void* d_out, int out_size, void* d_ws, size_t ws_size,
                              hipStream_t stream) {
    const float* h         = (const float*)d_in[0];
    const float* x         = (const float*)d_in[1];
    const int*   edgei     = (const int*)d_in[2];
    const int*   edgej     = (const int*)d_in[3];
    const float* node_attr = (const float*)d_in[4];
    const float* W_e1      = (const float*)d_in[5];
    const float* g_e       = (const float*)d_in[6];
    const float* be_e      = (const float*)d_in[7];
    const float* W_e2      = (const float*)d_in[8];
    const float* b_e2      = (const float*)d_in[9];
    const float* W_m       = (const float*)d_in[10];
    const float* b_m       = (const float*)d_in[11];
    const float* W_x1      = (const float*)d_in[12];
    const float* b_x1      = (const float*)d_in[13];
    const float* W_x2      = (const float*)d_in[14];
    const float* W_h1      = (const float*)d_in[15];
    const float* b_h1      = (const float*)d_in[16];
    const float* g_h       = (const float*)d_in[17];
    const float* be_h      = (const float*)d_in[18];
    const float* W_h2      = (const float*)d_in[19];
    const float* b_h2      = (const float*)d_in[20];

    float* ws     = (float*)d_ws;                 // total need ~3.1 MB
    float* sumE   = ws + 0;
    float* sumsqE = ws + 72;
    float* scE    = ws + 144;   // 80 slots
    float* shE    = ws + 224;   // 80
    float* sumH   = ws + 304;
    float* sumsqH = ws + 376;
    float* scH    = ws + 448;   // 80
    float* shH    = ws + 528;   // 80
    float* agg    = ws + 608;                    // B*N*4
    float* cnt    = agg + B_ * N_ * 4;           // B*N
    float* magg   = cnt + B_ * N_;               // B*N*72
    float* zh1    = magg + B_ * N_ * NHc;        // B*N*72
    // bf16 pre-converted images (16B aligned: 610912*4 bytes offset)
    unsigned short* wtE1g = (unsigned short*)(zh1 + B_ * N_ * NHc);
    unsigned short* wtH1g = wtE1g + CP * K1;
    unsigned short* wtE2g = wtH1g + CP * K1;
    unsigned short* wtX1g = wtE2g + CP * K2;
    unsigned short* wtH2g = wtX1g + CP * K2;
    unsigned short* hbf   = wtH2g + CP * K2;     // B*N*72 bf16

    float* out_h = (float*)d_out;
    float* out_x = out_h + B_ * N_ * NOc;
    float* out_m = out_x + B_ * N_ * 4;

    const int nzero = 608 + B_ * N_ * 4 + B_ * N_ + B_ * N_ * NHc;   // accumulators
    const int S1 = (TME * K1 + CP * K1) * 2 + (2 * CP) * 4 + 2 * TME * 4;
    const int S3 = (TME * K1 + CP * K1 + TME * K2) * 2
                 + (4 * CP + 2 * NHc + TME + 4 * TME) * 4 + 2 * TME * 4;
    const int S4 = (TMN * K1 + CP * K1) * 2 + (3 * CP) * 4;
    const int S6 = (TMN * K2 + CP * K2) * 2 + (3 * CP) * 4;

    // stage 0: zero accumulators + build bf16 operand images
    k_zero<<<(nzero + 255) / 256, 256, 0, stream>>>(ws, nzero);
    k_conv_wt<<<(CP * K1 + 255) / 256, 256, 0, stream>>>(W_e1, wtE1g, 2 * NIc + 2, NHc, K1);
    k_conv_wt<<<(CP * K1 + 255) / 256, 256, 0, stream>>>(W_h1, wtH1g, NIc + NHc + NAc, NHc, K1);
    k_conv_wt<<<(CP * K2 + 255) / 256, 256, 0, stream>>>(W_e2, wtE2g, NHc, NHc, K2);
    k_conv_wt<<<(CP * K2 + 255) / 256, 256, 0, stream>>>(W_x1, wtX1g, NHc, NHc, K2);
    k_conv_wt<<<(CP * K2 + 255) / 256, 256, 0, stream>>>(W_h2, wtH2g, NHc, NOc, K2);
    k_conv_h<<<(B_ * N_ * NIc + 255) / 256, 256, 0, stream>>>(h, hbf, B_ * N_ * NIc);

    // stage 1-2: edge BN statistics
    k_edge_stats<<<(B_ * E_) / TME, 256, S1, stream>>>(hbf, x, edgei, edgej, wtE1g, sumE, sumsqE);
    k_bn_finalize<<<1, 128, 0, stream>>>(sumE, sumsqE, g_e, be_e, scE, shE, (float)(B_ * E_));

    // stage 3: fused edge pipeline
    k_edge_main<<<(B_ * E_) / TME, 256, S3, stream>>>(hbf, x, edgei, edgej,
                                                      wtE1g, wtE2g, wtX1g,
                                                      b_e2, W_m, b_m, b_x1, W_x2,
                                                      scE, shE, out_m, magg, agg, cnt);

    // stage 4-6: node pipeline
    k_node_gemm1<<<(B_ * N_) / TMN, 256, S4, stream>>>(hbf, node_attr, magg, wtH1g, b_h1,
                                                       zh1, sumH, sumsqH);
    k_bn_finalize<<<1, 128, 0, stream>>>(sumH, sumsqH, g_h, be_h, scH, shH, (float)(B_ * N_));
    k_node_final<<<(B_ * N_) / TMN, 256, S6, stream>>>(zh1, scH, shH, wtH2g, b_h2,
                                                       h, x, agg, cnt, out_h, out_x);
}
